// GPGIN_23321672417876
// MI455X (gfx1250) — matrix operand order repaired and checked by reading, not compiled, not run on hardware
//
#include <hip/hip_runtime.h>
#include <math.h>

#define NN 20000
#define NE 320000
#define DD 128
#define HH 256
#define GG 64
#define LL 3
#define BB 64
#define EPSV 1e-5f

typedef __attribute__((ext_vector_type(16))) _Float16 v16h;
typedef __attribute__((ext_vector_type(8)))  float    v8f;

__device__ __forceinline__ float silu_f(float x){ return x * (1.0f/(1.0f + __expf(-x))); }

// non-returning global fp32 atomic add (STOREcnt path, no CAS loop)
__device__ __forceinline__ void gatomic_add(float* p, float v){
  asm volatile("global_atomic_add_f32 %0, %1, off" :: "v"(p), "v"(v) : "memory");
}

// Inverse of the CDNA5 16-bit A-matrix 16x32 lane layout (ISA 7.12.2):
// element (m, kk) of a 16x32 tile lives at (lane, i).
__device__ __forceinline__ void a_swz(int m, int kk, int& lane, int& i){
  int hf, j; const int p = kk & 1;
  if      (kk < 8)  { hf = 0; j = kk >> 1; }
  else if (kk < 16) { hf = 1; j = (kk - 8) >> 1; }
  else if (kk < 24) { hf = 0; j = 4 + ((kk - 16) >> 1); }
  else              { hf = 1; j = 4 + ((kk - 24) >> 1); }
  lane = hf*16 + m;
  i = 2*j + p;
}

// ---- weight swizzle: row-major f32 [K][N] (LL layers) -> fp16 B-operand tiles.
// B-matrix 32x16 layout: lane = ((k>>3)&1)*16 + n%16, i = (k&7) + ((k>>4)<<3)  (table order, matches A)
__global__ void k_swz_w(const float* __restrict__ src, _Float16* __restrict__ dst,
                        int K, int Nn, int total){
  int idx = blockIdx.x*blockDim.x + threadIdx.x;
  if (idx >= total) return;
  const int per = K*Nn;
  const int layer = idx / per, r = idx % per;
  const int k = r / Nn, n = r % Nn;
  const int kt = k >> 5, kk = k & 31;
  const int nt = n >> 4, nn = n & 15;
  const int lane = (((kk >> 3) & 1) << 4) + nn;
  const int i = (kk & 7) + ((kk >> 4) << 3);
  const size_t o = (size_t)layer*per + (((size_t)kt*(Nn >> 4) + nt)*32 + lane)*16 + i;
  dst[o] = (_Float16)src[idx];
}

__global__ void k_cnt(const int* __restrict__ batch, float* __restrict__ cnt){
  int i = blockIdx.x*blockDim.x + threadIdx.x;
  if (i < NN) gatomic_add(&cnt[batch[i]], 1.0f);
}

__global__ void k_init_h(const float* __restrict__ emb, const int* __restrict__ atype,
                         float* __restrict__ h){
  int i = blockIdx.x*blockDim.x + threadIdx.x;  // over NN*DD
  int n = i >> 7, d = i & 127;
  h[i] = emb[(size_t)atype[n]*DD + d];
}

// gaussian smearing, written directly in WMMA A-operand order per 16-edge block:
// layout [eblk][kchunk(2)][lane(32)][i(16)]
__global__ void k_gauss(const float* __restrict__ pos, const int* __restrict__ ei,
                        _Float16* __restrict__ ea0){
  int e = blockIdx.x*blockDim.x + threadIdx.x;
  if (e >= NE) return;
  const int r = ei[e], c = ei[NE + e];
  float dx = pos[r*3+0]-pos[c*3+0];
  float dy = pos[r*3+1]-pos[c*3+1];
  float dz = pos[r*3+2]-pos[c*3+2];
  float dist = sqrtf(dx*dx+dy*dy+dz*dz + 1e-12f);
  const float delta = 10.0f/(GG-1);
  const float coeff = -0.5f/(delta*delta);
  const int eblk = e >> 4, m = e & 15;
#pragma unroll 8
  for (int g = 0; g < GG; ++g){
    float t = dist - g*delta;
    float v = __expf(coeff*t*t);
    int lane, i;
    a_swz(m, g & 31, lane, i);
    ea0[(((size_t)eblk*2 + (g >> 5))*32 + lane)*16 + i] = (_Float16)v;
  }
}

// ---------------- fused edge MLP + message + scatter-add ----------------
__global__ __launch_bounds__(128) void k_edge(
    const _Float16* __restrict__ ea0, const _Float16* __restrict__ W0,
    const float* __restrict__ b0, const float* __restrict__ lng, const float* __restrict__ lnb,
    const _Float16* __restrict__ W1, const float* __restrict__ b1,
    const float* __restrict__ h, const int* __restrict__ ei, float* __restrict__ h0)
{
  __shared__ float sMid[16*264];
  __shared__ v16h  sMidHv[8*32];     // GEMM2 A operand, A-swizzled [kchunk][lane]
  __shared__ float sEa[16*DD];
  __shared__ float red1[16][8], red2[16][8];
  _Float16* sMidH = (_Float16*)sMidHv;

  const int t = threadIdx.x, lane = t & 31, wave = t >> 5;
  const int e0 = blockIdx.x * 16;
  const v16h* Av = (const v16h*)ea0 + (size_t)blockIdx.x*2*32;   // 2 k-chunks
  const v16h* B0v = (const v16h*)W0;                             // [kt][nt][lane]
  const v16h* B1v = (const v16h*)W1;
  __builtin_prefetch(W0, 0, 0);
  __builtin_prefetch(W1, 0, 0);

  // GEMM1: [16,64] x [64,256] -> sMid (fp32 + bias); A straight from global
  for (int nt = wave*4; nt < wave*4 + 4; ++nt){
    v8f acc = {};
#pragma unroll
    for (int kt = 0; kt < 2; ++kt){
      v16h a = Av[kt*32 + lane];
      v16h b = B0v[((size_t)kt*(HH/16) + nt)*32 + lane];
      acc = __builtin_amdgcn_wmma_f32_16x16x32_f16(false, a, false, b, (short)0, acc, false, false);
    }
    const int n = lane & 15, hf = lane >> 4;
    const float bias = b0[nt*16 + n];
#pragma unroll
    for (int r = 0; r < 8; ++r) sMid[(hf*8+r)*264 + nt*16 + n] = acc[r] + bias;
  }
  __syncthreads();

  // LayerNorm(256) + SiLU -> fp16, stored A-swizzled for GEMM2
  {
    const int r = t >> 3, s = t & 7;
    float s1 = 0.f, s2 = 0.f;
    for (int c = s*32; c < s*32+32; ++c){ float x = sMid[r*264+c]; s1 += x; s2 += x*x; }
    red1[r][s] = s1; red2[r][s] = s2;
    __syncthreads();
    float S1 = 0.f, S2 = 0.f;
#pragma unroll
    for (int q = 0; q < 8; ++q){ S1 += red1[r][q]; S2 += red2[r][q]; }
    const float mu = S1*(1.0f/HH);
    const float var = S2*(1.0f/HH) - mu*mu;
    const float rs = rsqrtf(var + EPSV);
    for (int c = s*32; c < s*32+32; ++c){
      float y = lng[c]*(sMid[r*264+c]-mu)*rs + lnb[c];
      int ln_, ii;
      a_swz(r, c & 31, ln_, ii);
      sMidH[((c >> 5)*32 + ln_)*16 + ii] = (_Float16)silu_f(y);
    }
  }
  __syncthreads();

  // GEMM2: [16,256] x [256,128] -> sEa (fp32 + bias)
  for (int nt = wave*2; nt < wave*2 + 2; ++nt){
    v8f acc = {};
#pragma unroll
    for (int kt = 0; kt < 8; ++kt){
      v16h a = sMidHv[kt*32 + lane];
      v16h b = B1v[((size_t)kt*(DD/16) + nt)*32 + lane];
      acc = __builtin_amdgcn_wmma_f32_16x16x32_f16(false, a, false, b, (short)0, acc, false, false);
    }
    const int n = lane & 15, hf = lane >> 4;
    const float bias = b1[nt*16 + n];
#pragma unroll
    for (int r = 0; r < 8; ++r) sEa[(hf*8+r)*DD + nt*16 + n] = acc[r] + bias;
  }
  __syncthreads();

  // message = h[row] * ea ; scatter-add into h0[col]
  {
    const int el = t >> 3, dg = t & 7;
    const int e = e0 + el;
    const int rsrc = ei[e];
    const int cdst = ei[NE + e];
#pragma unroll
    for (int dd = 0; dd < 16; ++dd){
      const int d = dg*16 + dd;
      float m = h[(size_t)rsrc*DD + d] * sEa[el*DD + d];
      gatomic_add(&h0[(size_t)cdst*DD + d], m);
    }
  }
}

// ---------------- fused node MLP (+ optional final LN/SiLU + graphnorm mean accum) ----------------
__global__ __launch_bounds__(128) void k_node(
    const float* __restrict__ h0, float* __restrict__ h, const float* __restrict__ calpha,
    const _Float16* __restrict__ W0, const float* __restrict__ b0,
    const float* __restrict__ lng, const float* __restrict__ lnb,
    const _Float16* __restrict__ W1, const float* __restrict__ b1,
    const float* __restrict__ lnfg, const float* __restrict__ lnfb,
    const float* __restrict__ alphag, int layer, int last,
    const int* __restrict__ batch, float* __restrict__ gsum, float* __restrict__ hcand)
{
  __shared__ v16h  sZv[4*32];        // GEMM1 A operand, A-swizzled (16x128)
  __shared__ float sMid[16*264];
  __shared__ v16h  sMidHv[8*32];     // GEMM2 A operand, A-swizzled (16x256)
  __shared__ float sOut[16*DD];
  __shared__ float red1[16][8], red2[16][8];
  _Float16* sZ = (_Float16*)sZv;
  _Float16* sMidH = (_Float16*)sMidHv;

  const int t = threadIdx.x, lane = t & 31, wave = t >> 5;
  const int n0 = blockIdx.x * 16;
  const v16h* B0v = (const v16h*)W0;
  const v16h* B1v = (const v16h*)W1;
  __builtin_prefetch(W0, 0, 0);

  // z = conv_alpha * h0 + h  -> fp16, A-swizzled
  {
    const int nl = t >> 3, dg = t & 7;
    const size_t base = (size_t)(n0+nl)*DD;
#pragma unroll
    for (int dd = 0; dd < 16; ++dd){
      const int d = dg*16 + dd;
      float z = calpha[d]*h0[base + d] + h[base + d];
      int ln_, ii;
      a_swz(nl, d & 31, ln_, ii);
      sZ[((d >> 5)*32 + ln_)*16 + ii] = (_Float16)z;
    }
  }
  __syncthreads();

  // GEMM1: [16,128] x [128,256]
  for (int nt = wave*4; nt < wave*4 + 4; ++nt){
    v8f acc = {};
#pragma unroll
    for (int kt = 0; kt < 4; ++kt){
      v16h a = sZv[kt*32 + lane];
      v16h b = B0v[((size_t)kt*(HH/16) + nt)*32 + lane];
      acc = __builtin_amdgcn_wmma_f32_16x16x32_f16(false, a, false, b, (short)0, acc, false, false);
    }
    const int n = lane & 15, hf = lane >> 4;
    const float bias = b0[nt*16 + n];
#pragma unroll
    for (int r = 0; r < 8; ++r) sMid[(hf*8+r)*264 + nt*16 + n] = acc[r] + bias;
  }
  __syncthreads();

  // LayerNorm(256) + SiLU -> fp16, A-swizzled
  {
    const int r = t >> 3, s = t & 7;
    float s1 = 0.f, s2 = 0.f;
    for (int c = s*32; c < s*32+32; ++c){ float x = sMid[r*264+c]; s1 += x; s2 += x*x; }
    red1[r][s] = s1; red2[r][s] = s2;
    __syncthreads();
    float S1 = 0.f, S2 = 0.f;
#pragma unroll
    for (int q = 0; q < 8; ++q){ S1 += red1[r][q]; S2 += red2[r][q]; }
    const float mu = S1*(1.0f/HH);
    const float var = S2*(1.0f/HH) - mu*mu;
    const float rs = rsqrtf(var + EPSV);
    for (int c = s*32; c < s*32+32; ++c){
      float y = lng[c]*(sMid[r*264+c]-mu)*rs + lnb[c];
      int ln_, ii;
      a_swz(r, c & 31, ln_, ii);
      sMidH[((c >> 5)*32 + ln_)*16 + ii] = (_Float16)silu_f(y);
    }
  }
  __syncthreads();

  // GEMM2: [16,256] x [256,128]
  for (int nt = wave*2; nt < wave*2 + 2; ++nt){
    v8f acc = {};
#pragma unroll
    for (int kt = 0; kt < 8; ++kt){
      v16h a = sMidHv[kt*32 + lane];
      v16h b = B1v[((size_t)kt*(DD/16) + nt)*32 + lane];
      acc = __builtin_amdgcn_wmma_f32_16x16x32_f16(false, a, false, b, (short)0, acc, false, false);
    }
    const int n = lane & 15, hf = lane >> 4;
    const float bias = b1[nt*16 + n];
#pragma unroll
    for (int r = 0; r < 8; ++r) sOut[(hf*8+r)*DD + nt*16 + n] = acc[r] + bias;
  }
  __syncthreads();

  if (last){  // uniform branch (kernel arg) -> safe
    const int nl = t >> 3, dg = t & 7;
    const size_t base = (size_t)(n0+nl)*DD;
    const float a = alphag[layer];
#pragma unroll
    for (int dd = 0; dd < 16; ++dd){
      const int d = dg*16 + dd;
      h[base + d] = h[base + d] + a*sOut[nl*DD + d];
    }
  } else {
    // final LayerNorm(128) + SiLU -> hcand, accumulate graphnorm mean
    const int r = t >> 3, s = t & 7;
    float s1 = 0.f, s2 = 0.f;
    for (int c = s*16; c < s*16+16; ++c){ float x = sOut[r*DD+c]; s1 += x; s2 += x*x; }
    red1[r][s] = s1; red2[r][s] = s2;
    __syncthreads();
    float S1 = 0.f, S2 = 0.f;
#pragma unroll
    for (int q = 0; q < 8; ++q){ S1 += red1[r][q]; S2 += red2[r][q]; }
    const float mu = S1*(1.0f/DD);
    const float var = S2*(1.0f/DD) - mu*mu;
    const float rs = rsqrtf(var + EPSV);
    const int bidx = batch[n0 + r];
    const size_t base = (size_t)(n0+r)*DD;
    for (int c = s*16; c < s*16+16; ++c){
      float y = lnfg[c]*(sOut[r*DD+c]-mu)*rs + lnfb[c];
      float hc = silu_f(y);
      hcand[base + c] = hc;
      gatomic_add(&gsum[bidx*DD + c], hc);
    }
  }
}

__global__ void k_gn_var(const float* __restrict__ gsum, float* __restrict__ hcand,
                         float* __restrict__ gssq, const float* __restrict__ cnt,
                         const int* __restrict__ batch, const float* __restrict__ ms){
  int i = blockIdx.x*blockDim.x + threadIdx.x;  // over NN*DD
  int n = i >> 7, d = i & 127;
  int b = batch[n];
  float c = fmaxf(cnt[b], 1.0f);
  float mean = gsum[b*DD + d] / c;
  float sub = hcand[i] - ms[d]*mean;
  hcand[i] = sub;
  gatomic_add(&gssq[b*DD + d], sub*sub);
}

__global__ void k_gn_fin(const float* __restrict__ hcand, const float* __restrict__ gssq,
                         const float* __restrict__ cnt, const int* __restrict__ batch,
                         const float* __restrict__ w, const float* __restrict__ bb,
                         const float* __restrict__ alphag, int layer, float* __restrict__ h){
  int i = blockIdx.x*blockDim.x + threadIdx.x;  // over NN*DD
  int n = i >> 7, d = i & 127;
  int b = batch[n];
  float c = fmaxf(cnt[b], 1.0f);
  float var = gssq[b*DD + d] / c;
  float gn = w[d]*hcand[i]*rsqrtf(var + EPSV) + bb[d];
  h[i] = h[i] + alphag[layer]*silu_f(gn);
}

__global__ void k_readout(const float* __restrict__ h, const int* __restrict__ batch,
                          float* __restrict__ osum){
  int n = blockIdx.x*blockDim.x + threadIdx.x;
  if (n >= NN) return;
  float s = 0.f;
  const size_t base = (size_t)n*DD;
#pragma unroll 8
  for (int d = 0; d < DD; ++d) s += h[base + d];
  gatomic_add(&osum[batch[n]], s);
}

__global__ void k_final(const float* __restrict__ osum, const float* __restrict__ cnt,
                        float* __restrict__ out){
  int b = threadIdx.x;
  if (b < BB) out[b] = osum[b] / (fmaxf(cnt[b], 1.0f) * (float)DD);
}

extern "C" void kernel_launch(void* const* d_in, const int* in_sizes, int n_in,
                              void* d_out, int out_size, void* d_ws, size_t ws_size,
                              hipStream_t stream){
  (void)in_sizes; (void)n_in; (void)out_size; (void)ws_size;
  const float* emb    = (const float*)d_in[0];
  const float* alphag = (const float*)d_in[1];
  const float* calpha = (const float*)d_in[2];
  const float* eW0f   = (const float*)d_in[3];
  const float* eb0    = (const float*)d_in[4];
  const float* elng   = (const float*)d_in[5];
  const float* elnb   = (const float*)d_in[6];
  const float* eW1f   = (const float*)d_in[7];
  const float* eb1    = (const float*)d_in[8];
  const float* nW0f   = (const float*)d_in[9];
  const float* nb0    = (const float*)d_in[10];
  const float* nlng   = (const float*)d_in[11];
  const float* nlnb   = (const float*)d_in[12];
  const float* nW1f   = (const float*)d_in[13];
  const float* nb1    = (const float*)d_in[14];
  const float* lnfg   = (const float*)d_in[15];
  const float* lnfb   = (const float*)d_in[16];
  const float* gnw    = (const float*)d_in[17];
  const float* gnb    = (const float*)d_in[18];
  const float* gnms   = (const float*)d_in[19];
  const float* pos    = (const float*)d_in[20];
  const int*   atype  = (const int*)d_in[21];
  const int*   eidx   = (const int*)d_in[22];
  const int*   batch  = (const int*)d_in[23];
  float* out = (float*)d_out;

  char* wp = (char*)d_ws;
  auto carve = [&](size_t bytes)->void*{
    void* p = (void*)wp; wp += (bytes + 255) & ~(size_t)255; return p;
  };
  _Float16* ea0 = (_Float16*)carve((size_t)NE*GG*sizeof(_Float16));
  _Float16* wE0 = (_Float16*)carve((size_t)LL*GG*HH*sizeof(_Float16));
  _Float16* wE1 = (_Float16*)carve((size_t)LL*HH*DD*sizeof(_Float16));
  _Float16* wN0 = (_Float16*)carve((size_t)LL*DD*HH*sizeof(_Float16));
  _Float16* wN1 = (_Float16*)carve((size_t)LL*HH*DD*sizeof(_Float16));
  float* h     = (float*)carve((size_t)NN*DD*4);
  float* h0    = (float*)carve((size_t)NN*DD*4);
  float* hcand = (float*)carve((size_t)NN*DD*4);
  float* gsum  = (float*)carve((size_t)BB*DD*4);
  float* gssq  = (float*)carve((size_t)BB*DD*4);
  float* cnt   = (float*)carve((size_t)BB*4);
  float* osum  = (float*)carve((size_t)BB*4);

  hipMemsetAsync(cnt, 0, BB*4, stream);
  k_cnt<<<(NN+255)/256, 256, 0, stream>>>(batch, cnt);
  k_init_h<<<(NN*DD)/256, 256, 0, stream>>>(emb, atype, h);
  k_swz_w<<<(LL*GG*HH+255)/256, 256, 0, stream>>>(eW0f, wE0, GG, HH, LL*GG*HH);
  k_swz_w<<<(LL*HH*DD+255)/256, 256, 0, stream>>>(eW1f, wE1, HH, DD, LL*HH*DD);
  k_swz_w<<<(LL*DD*HH+255)/256, 256, 0, stream>>>(nW0f, wN0, DD, HH, LL*DD*HH);
  k_swz_w<<<(LL*HH*DD+255)/256, 256, 0, stream>>>(nW1f, wN1, HH, DD, LL*HH*DD);
  k_gauss<<<(NE+255)/256, 256, 0, stream>>>(pos, eidx, ea0);

  for (int i = 0; i < LL; ++i){
    hipMemsetAsync(h0, 0, (size_t)NN*DD*4, stream);
    k_edge<<<NE/16, 128, 0, stream>>>(ea0, wE0 + (size_t)i*GG*HH, eb0 + i*HH,
                                      elng + i*HH, elnb + i*HH,
                                      wE1 + (size_t)i*HH*DD, eb1 + i*DD,
                                      h, eidx, h0);
    const int last = (i == LL-1) ? 1 : 0;
    if (!last){
      hipMemsetAsync(gsum, 0, BB*DD*4, stream);
      hipMemsetAsync(gssq, 0, BB*DD*4, stream);
    }
    k_node<<<NN/16, 128, 0, stream>>>(h0, h, calpha + i*DD,
                                      wN0 + (size_t)i*DD*HH, nb0 + i*HH,
                                      nlng + i*HH, nlnb + i*HH,
                                      wN1 + (size_t)i*HH*DD, nb1 + i*DD,
                                      lnfg + i*DD, lnfb + i*DD,
                                      alphag, i, last, batch, gsum, hcand);
    if (!last){
      k_gn_var<<<(NN*DD)/256, 256, 0, stream>>>(gsum, hcand, gssq, cnt, batch, gnms + i*DD);
      k_gn_fin<<<(NN*DD)/256, 256, 0, stream>>>(hcand, gssq, cnt, batch,
                                                gnw + i*DD, gnb + i*DD, alphag, i, h);
    }
  }

  hipMemsetAsync(osum, 0, BB*4, stream);
  k_readout<<<(NN+255)/256, 256, 0, stream>>>(h, batch, osum);
  k_final<<<1, 64, 0, stream>>>(osum, cnt, out);
}
